// NGCF_29703993819989
// MI455X (gfx1250) — matrix operand order repaired and checked by reading, not compiled, not run on hardware
//
#include <hip/hip_runtime.h>

typedef __attribute__((ext_vector_type(16))) _Float16 v16h;
typedef __attribute__((ext_vector_type(8)))  float    v8f;

#define EMB    64
#define CONCAT 256
#define SLOPE  0.2f

// ---- fragment layout helpers (per CDNA5 ISA 7.12.2, wave32) ----------------
// 16-bit A matrix 16x32: lane l holds row m = l&15; lane-group g = l>>4.
// VGPR j/2 pairs: g==0 -> K in {0..7, 16..23}; g==1 -> +8.
__device__ __forceinline__ int a_klocal(int j, int grp) {
    int vg = j >> 1, pos = j & 1;
    int k = (vg < 4) ? (2 * vg) : (16 + 2 * (vg - 4));
    return k + pos + 8 * grp;
}

// A fragment from row-major f32 source (stride ld), rows m0..m0+15, cols kbase..kbase+31
__device__ __forceinline__ v16h load_a_f32(const float* __restrict__ src, int ld,
                                           int m0, int kbase, int lane) {
    int m = m0 + (lane & 15);
    int grp = lane >> 4;
    const float* p = src + (long long)m * ld + kbase;
    v16h a;
#pragma unroll
    for (int j = 0; j < 16; ++j) a[j] = (_Float16)p[a_klocal(j, grp)];
    return a;
}

// A fragment = elementwise product of two f32 sources (for the bi-interaction term)
__device__ __forceinline__ v16h load_a_prod(const float* __restrict__ s0,
                                            const float* __restrict__ s1, int ld,
                                            int m0, int kbase, int lane) {
    int m = m0 + (lane & 15);
    int grp = lane >> 4;
    long long base = (long long)m * ld + kbase;
    v16h a;
#pragma unroll
    for (int j = 0; j < 16; ++j) {
        int k = a_klocal(j, grp);
        a[j] = (_Float16)(s0[base + k] * s1[base + k]);
    }
    return a;
}

// A fragment from row-major f16 source
__device__ __forceinline__ v16h load_a_f16(const _Float16* __restrict__ src, int ld,
                                           int m0, int kbase, int lane) {
    int m = m0 + (lane & 15);
    int grp = lane >> 4;
    const _Float16* p = src + (long long)m * ld + kbase;
    v16h a;
#pragma unroll
    for (int j = 0; j < 16; ++j) a[j] = p[a_klocal(j, grp)];
    return a;
}

// ---- pre-swizzled B fragments in LDS ---------------------------------------
// Fragment f (= kh*4 + ntile) stored as [32 lanes][16 halfs], lane data contiguous.
// B 32x16 (KxN) layout: lane l -> column n = n0 + (l&15); element j -> K = kbase + 8*(l>>4) + (j&7) + 16*(j>>3).
// Staging permutation: LDS[((f*32 + lane)<<4) + j] = W[(kh*32 + 8*(lane>>4) + (j&7) + 16*(j>>3))*64 + nt*16 + (lane&15)]
__device__ __forceinline__ void stage_b_frags(const float* __restrict__ W,
                                              _Float16* __restrict__ sW,
                                              int nfrag /* kh count * 4 */) {
    int total = nfrag * 32 * 16;
    for (int i = threadIdx.x; i < total; i += 256) {
        int j = i & 15;
        int lane = (i >> 4) & 31;
        int f = i >> 9;
        int kh = f >> 2, nt = f & 3;
        int K = kh * 32 + 8 * (lane >> 4) + (j & 7) + 16 * (j >> 3);
        int Nc = nt * 16 + (lane & 15);
        sW[i] = (_Float16)W[K * EMB + Nc];
    }
}

// one aligned 32-byte LDS read -> 2x ds_load_b128
__device__ __forceinline__ v16h load_b_frag(const _Float16* __restrict__ sW,
                                            int f, int lane) {
    return *(const v16h*)(sW + (((f << 5) + lane) << 4));
}

#define WMMA_F16(A, B, C) \
    __builtin_amdgcn_wmma_f32_16x16x32_f16(false, (A), false, (B), (short)0, (C), false, false)

// ---- kernel 1: seed concat slice 0 with input embeddings (f32 -> f16) ------
__global__ __launch_bounds__(256) void ngcf_cvt0_kernel(
    const float* __restrict__ ego, _Float16* __restrict__ allemb, int total) {
    int t = blockIdx.x * 256 + threadIdx.x;
    if (t >= total) return;
    int node = t >> 6, d = t & 63;
    allemb[(long long)node * CONCAT + d] = (_Float16)ego[t];
}

// ---- kernel 2: edge-parallel SpMM scatter (A_hat @ x), L2-resident atomics -
__global__ __launch_bounds__(256) void ngcf_spmm_kernel(
    const int* __restrict__ row, const int* __restrict__ col,
    const float* __restrict__ val, const float* __restrict__ x,
    float* __restrict__ side, int nedges) {
    long long t = (long long)blockIdx.x * 256 + threadIdx.x;
    int e = (int)(t >> 4);
    if (e >= nedges) return;
    int c = ((int)t & 15) * 4;
    int r = row[e], s = col[e];
    float v = val[e];
    const float4 xv = *(const float4*)(x + (long long)s * EMB + c);
    float* dst = side + (long long)r * EMB + c;
    atomicAdd(dst + 0, v * xv.x);
    atomicAdd(dst + 1, v * xv.y);
    atomicAdd(dst + 2, v * xv.z);
    atomicAdd(dst + 3, v * xv.w);
}

// ---- kernel 3: fused dual-GEMM + bias + leaky-relu + row-normalize ---------
// ego_next = leaky(side@Wgc + (ego*side)@Wbi + bgc + bbi)  (f32, unnormalized)
// allemb[:, slice*64 : slice*64+64] = normalize(ego_next)  (f16)
__global__ __launch_bounds__(256) void ngcf_layer_kernel(
    const float* __restrict__ side, const float* __restrict__ ego,
    const float* __restrict__ Wgc, const float* __restrict__ bgc,
    const float* __restrict__ Wbi, const float* __restrict__ bbi,
    float* __restrict__ ego_out, _Float16* __restrict__ allemb,
    int slice, int ntiles) {
    __shared__ __align__(32) _Float16 sWgc[8 * 32 * 16];   // 8 fragments, pre-swizzled
    __shared__ __align__(32) _Float16 sWbi[8 * 32 * 16];
    stage_b_frags(Wgc, sWgc, 8);
    stage_b_frags(Wbi, sWbi, 8);
    __syncthreads();

    int wave = threadIdx.x >> 5;
    int lane = threadIdx.x & 31;
    int tile = blockIdx.x * 8 + wave;
    if (tile >= ntiles) return;            // whole-wave exit: EXEC stays all-1 for WMMA
    int m0 = tile * 16;
    int grp = lane >> 4;
    int nl = lane & 15;

    v16h a1k0 = load_a_f32(side, EMB, m0, 0, lane);
    v16h a1k1 = load_a_f32(side, EMB, m0, 32, lane);
    v16h a2k0 = load_a_prod(side, ego, EMB, m0, 0, lane);
    v16h a2k1 = load_a_prod(side, ego, EMB, m0, 32, lane);

    float e[4][8];
    float ss[8];
#pragma unroll
    for (int r = 0; r < 8; ++r) ss[r] = 0.f;

#pragma unroll
    for (int nt = 0; nt < 4; ++nt) {
        v8f acc = {};
        acc = WMMA_F16(a1k0, load_b_frag(sWgc, 0 + nt, lane), acc);   // kh=0
        acc = WMMA_F16(a1k1, load_b_frag(sWgc, 4 + nt, lane), acc);   // kh=1
        acc = WMMA_F16(a2k0, load_b_frag(sWbi, 0 + nt, lane), acc);
        acc = WMMA_F16(a2k1, load_b_frag(sWbi, 4 + nt, lane), acc);
        int n0 = nt * 16;
        float bias = bgc[n0 + nl] + bbi[n0 + nl];
#pragma unroll
        for (int r = 0; r < 8; ++r) {       // C/D layout: row m0+r+8*grp, col n0+nl
            float x = acc[r] + bias;
            x = (x >= 0.f) ? x : SLOPE * x; // leaky relu
            e[nt][r] = x;
            ss[r] += x * x;
            ego_out[(long long)(m0 + r + 8 * grp) * EMB + n0 + nl] = x;
        }
    }
    // L2 row norm: each row's 16 cols-per-ntile live across the 16-lane half-wave
#pragma unroll
    for (int r = 0; r < 8; ++r) {
        float s = ss[r];
        s += __shfl_xor(s, 1, 32);
        s += __shfl_xor(s, 2, 32);
        s += __shfl_xor(s, 4, 32);
        s += __shfl_xor(s, 8, 32);
        ss[r] = 1.0f / fmaxf(sqrtf(s), 1e-12f);
    }
#pragma unroll
    for (int nt = 0; nt < 4; ++nt)
#pragma unroll
        for (int r = 0; r < 8; ++r) {
            long long m = m0 + r + 8 * grp;
            allemb[m * CONCAT + slice * EMB + nt * 16 + nl] =
                (_Float16)(e[nt][r] * ss[r]);
        }
}

// ---- kernel 4: final projection  out = allemb(f16) @ projW + projb --------
__global__ __launch_bounds__(256) void ngcf_proj_kernel(
    const _Float16* __restrict__ allemb, const float* __restrict__ projW,
    const float* __restrict__ projb, float* __restrict__ out, int ntiles) {
    __shared__ __align__(32) _Float16 sW[32 * 32 * 16];  // 32 fragments (kh=0..7, nt=0..3), 32 KB
    stage_b_frags(projW, sW, 32);
    __syncthreads();

    int wave = threadIdx.x >> 5;
    int lane = threadIdx.x & 31;
    int tile = blockIdx.x * 8 + wave;
    if (tile >= ntiles) return;
    int m0 = tile * 16;
    int grp = lane >> 4;
    int nl = lane & 15;

    v16h afr[8];
#pragma unroll
    for (int kh = 0; kh < 8; ++kh) afr[kh] = load_a_f16(allemb, CONCAT, m0, kh * 32, lane);

#pragma unroll
    for (int nt = 0; nt < 4; ++nt) {
        v8f acc = {};
#pragma unroll
        for (int kh = 0; kh < 8; ++kh)
            acc = WMMA_F16(afr[kh], load_b_frag(sW, kh * 4 + nt, lane), acc);
        float b = projb[nt * 16 + nl];
#pragma unroll
        for (int r = 0; r < 8; ++r)
            out[(long long)(m0 + r + 8 * grp) * EMB + nt * 16 + nl] = acc[r] + b;
    }
}

// ---------------------------------------------------------------------------
extern "C" void kernel_launch(void* const* d_in, const int* in_sizes, int n_in,
                              void* d_out, int out_size, void* d_ws, size_t ws_size,
                              hipStream_t stream) {
    (void)n_in; (void)out_size; (void)ws_size;
    const float* ego_in  = (const float*)d_in[0];
    const float* adj_val = (const float*)d_in[1];
    const float* W_gc    = (const float*)d_in[2];
    const float* b_gc    = (const float*)d_in[3];
    const float* W_bi    = (const float*)d_in[4];
    const float* b_bi    = (const float*)d_in[5];
    const float* proj_W  = (const float*)d_in[6];
    const float* proj_b  = (const float*)d_in[7];
    const int*   adj_row = (const int*)d_in[8];
    const int*   adj_col = (const int*)d_in[9];

    const int N = in_sizes[0] / EMB;       // 200000
    const int E = in_sizes[8];             // 3200000
    const int ntiles = N / 16;             // 12500 (exact)

    // workspace layout
    float*    side  = (float*)d_ws;
    float*    egoA  = side + (size_t)N * EMB;
    float*    egoB  = egoA + (size_t)N * EMB;
    _Float16* allemb = (_Float16*)(egoB + (size_t)N * EMB);

    dim3 blk(256);

    // concat slice 0 = input embeddings
    ngcf_cvt0_kernel<<<(N * EMB + 255) / 256, blk, 0, stream>>>(ego_in, allemb, N * EMB);

    const float* ego_cur = ego_in;
    float* ego_bufs[2] = {egoA, egoB};
    for (int k = 0; k < 3; ++k) {
        hipMemsetAsync(side, 0, (size_t)N * EMB * sizeof(float), stream);
        unsigned int spmm_grid = (unsigned int)(((long long)E * 16 + 255) / 256);
        ngcf_spmm_kernel<<<spmm_grid, blk, 0, stream>>>(adj_row, adj_col, adj_val,
                                                        ego_cur, side, E);
        float* ego_next = ego_bufs[k & 1];
        ngcf_layer_kernel<<<(ntiles + 7) / 8, blk, 0, stream>>>(
            side, ego_cur,
            W_gc + (size_t)k * EMB * EMB, b_gc + (size_t)k * EMB,
            W_bi + (size_t)k * EMB * EMB, b_bi + (size_t)k * EMB,
            ego_next, allemb, k + 1, ntiles);
        ego_cur = ego_next;
    }

    ngcf_proj_kernel<<<(ntiles + 7) / 8, blk, 0, stream>>>(allemb, proj_W, proj_b,
                                                           (float*)d_out, ntiles);
}
